// AutoregressiveLSTM_2929167696587
// MI455X (gfx1250) — matrix operand order repaired and checked by reading, not compiled, not run on hardware
//
#include <hip/hip_runtime.h>
#include <hip/hip_bf16.h>

// ---------------------------------------------------------------------------
// Fused 2-layer LSTM LM for MI455X (gfx1250), wave32 + bf16 WMMA.
//   grid = 256 WGs x 256 threads; each WG owns 16 sequences for all 127 steps.
//   gates[16,1024] = act[16,K] @ Wcat^T via v_wmma_f32_16x16x32_bf16.
//   Wave w owns gate N-tiles {w+8t}: i/f/g/o of the same hidden units stay in
//   one wave -> LSTM cell fully in registers (c-state never leaves VGPRs).
//   Weights (~1.8 MB bf16) stay L2-resident and are streamed per step; an
//   opaque SGPR zero-offset inside the time loop stops LICM from hoisting the
//   weight loads (which would spill to scratch) while keeping the pointers'
//   global address space (global_load_b128, LOADcnt-only tracking).
// ---------------------------------------------------------------------------

typedef __attribute__((ext_vector_type(16))) __bf16 v16bf;
typedef __attribute__((ext_vector_type(8)))  __bf16 v8bf;
typedef __attribute__((ext_vector_type(8)))  float  v8f;

#define N_SEQ 4096
#define S_LEN 128
#define T_LEN 127
#define VOCAB 36
#define EDIM  128
#define HDIM  256
#define ACT_W 640            // [ x(128) | h0(256) | h1(256) ] bf16 per row
#define K0    384            // layer0 GEMM K = E + H
#define K1    512            // layer1 GEMM K = H + H

// workspace layout (bytes)
#define OFF_WCAT0 0u         // [1024][384] bf16  = 786432
#define OFF_WCAT1 786432u    // [1024][512] bf16  = 1048576
#define OFF_FCW   1835008u   // [48][256]  bf16   = 24576 (rows 36..47 zero)
#define OFF_EMB   1859584u   // [36][128]  bf16   = 9216
#define OFF_B0    1868800u   // [1024] f32 (b_ih0+b_hh0)
#define OFF_B1    1872896u   // [1024] f32
#define WS_NEED   1876992u

__device__ __forceinline__ unsigned short f2bf(float f) {
    unsigned int u = __float_as_uint(f);
    u += 0x7FFFu + ((u >> 16) & 1u);          // round-to-nearest-even
    return (unsigned short)(u >> 16);
}
__device__ __forceinline__ float sigm(float x) { return 1.0f / (1.0f + __expf(-x)); }

// ---------------------------------------------------------------------------
// Prep: bf16-convert + concatenate weights, combine biases.
// ---------------------------------------------------------------------------
__global__ void prep_kernel(const float* __restrict__ Wih0, const float* __restrict__ Whh0,
                            const float* __restrict__ bih0, const float* __restrict__ bhh0,
                            const float* __restrict__ Wih1, const float* __restrict__ Whh1,
                            const float* __restrict__ bih1, const float* __restrict__ bhh1,
                            const float* __restrict__ fcW,  const float* __restrict__ emb,
                            unsigned char* __restrict__ ws)
{
    unsigned int i = blockIdx.x * 256u + threadIdx.x;
    unsigned short* wc0 = (unsigned short*)(ws + OFF_WCAT0);
    unsigned short* wc1 = (unsigned short*)(ws + OFF_WCAT1);
    unsigned short* fcw = (unsigned short*)(ws + OFF_FCW);
    unsigned short* eb  = (unsigned short*)(ws + OFF_EMB);
    float* b0 = (float*)(ws + OFF_B0);
    float* b1 = (float*)(ws + OFF_B1);

    if (i < 393216u) {                                   // Wcat0 = [Wih0 | Whh0]
        unsigned int g = i / 384u, k = i % 384u;
        float v = (k < 128u) ? Wih0[g * 128u + k] : Whh0[g * 256u + (k - 128u)];
        wc0[i] = f2bf(v);
    } else if ((i -= 393216u) < 524288u) {               // Wcat1 = [Wih1 | Whh1]
        unsigned int g = i / 512u, k = i % 512u;
        float v = (k < 256u) ? Wih1[g * 256u + k] : Whh1[g * 256u + (k - 256u)];
        wc1[i] = f2bf(v);
    } else if ((i -= 524288u) < 12288u) {                // fcW padded to 48 rows
        unsigned int r = i / 256u;
        fcw[i] = f2bf((r < 36u) ? fcW[i] : 0.0f);
    } else if ((i -= 12288u) < 4608u) {                  // embedding table
        eb[i] = f2bf(emb[i]);
    } else if ((i -= 4608u) < 1024u) {
        b0[i] = bih0[i] + bhh0[i];
    } else if ((i -= 1024u) < 1024u) {
        b1[i] = bih1[i] + bhh1[i];
    }
}

// ---------------------------------------------------------------------------
// Main fused kernel.
// ---------------------------------------------------------------------------
__global__ __launch_bounds__(256)
void lstm_kernel(const int* __restrict__ x,
                 const unsigned short* __restrict__ wc0,
                 const unsigned short* __restrict__ wc1,
                 const unsigned short* __restrict__ fcw,
                 const unsigned short* __restrict__ embb,
                 const float* __restrict__ b0g,
                 const float* __restrict__ b1g,
                 const float* __restrict__ fcb,
                 float* __restrict__ out)
{
    __shared__ unsigned short s_emb[VOCAB * EDIM];   // bf16 table (9 KB)
    __shared__ unsigned char  s_tok[16 * S_LEN];     // tokens (2 KB)
    __shared__ unsigned short s_act[16 * ACT_W];     // [x|h0|h1] bf16 (20 KB)
    __shared__ float          s_lg[16 * 48];         // logits (3 KB)
    __shared__ float          s_mx[16], s_lse[16], s_len[16], s_nll[16];

    const int tid   = threadIdx.x;
    const int wg    = blockIdx.x;         // 0..255 -> rows [wg*16, wg*16+16)
    const int w     = tid >> 5;           // wave 0..7
    const int lane  = tid & 31;
    const int lhalf = lane & 15;          // WMMA N / M-row selector
    const int lhi   = lane >> 4;          // half-wave selector

    for (int i = tid; i < VOCAB * EDIM; i += 256) s_emb[i] = embb[i];
    for (int i = tid; i < 16 * S_LEN; i += 256) {
        int r = i >> 7, t = i & 127;
        s_tok[i] = (unsigned char)x[(wg * 16 + r) * S_LEN + t];
    }
    for (int i = tid; i < 16 * 512; i += 256) {      // zero h0|h1
        int r = i >> 9, c = i & 511;
        s_act[r * ACT_W + 128 + c] = 0;
    }
    if (tid < 16) s_nll[tid] = 0.0f;
    __syncthreads();
    if (tid < 16) {
        int cnt = 0;
        for (int t = 0; t < S_LEN; ++t) cnt += (s_tok[tid * S_LEN + t] != 0);
        s_len[tid] = (float)cnt;
    }

    // per-lane gate biases: acc tile t covers gate unit (w+8t)*16 + lhalf
    float b0r[8], b1r[8];
#pragma unroll
    for (int t = 0; t < 8; ++t) {
        int u = (w + 8 * t) * 16 + lhalf;
        b0r[t] = b0g[u];
        b1r[t] = b1g[u];
    }
    const int fccol = w * 16 + lhalf;
    float fb = 0.0f;
    if (w < 3 && fccol < VOCAB) fb = fcb[fccol];

    float c0r[2][8], c1r[2][8];
#pragma unroll
    for (int g = 0; g < 2; ++g)
#pragma unroll
        for (int v = 0; v < 8; ++v) { c0r[g][v] = 0.0f; c1r[g][v] = 0.0f; }

    __syncthreads();

    for (int t = 0; t < T_LEN; ++t) {
        // Opaque zero offsets (SGPR) regenerated each iteration: the weight
        // load addresses are no longer provably loop-invariant, so LICM can't
        // hoist + spill them, while the pointers themselves keep their global
        // address space -> global_load_b128, LOADcnt-only.
        unsigned int zo0 = 0u, zo1 = 0u, zof = 0u;
        asm volatile("" : "+s"(zo0), "+s"(zo1), "+s"(zof));

        // -------- embed x_t into act[:,0:128] (16B per thread) --------
        {
            int r  = tid >> 4;
            int c8 = (tid & 15) * 8;
            int tk = s_tok[r * S_LEN + t];
            const unsigned short* src = &s_emb[tk * EDIM + c8];
            unsigned short*       dst = &s_act[r * ACT_W + c8];
#pragma unroll
            for (int q = 0; q < 8; ++q) dst[q] = src[q];
        }
        __syncthreads();

        v8f acc[8];

        // -------- layer 0: gates = [x|h0] @ Wcat0^T  (K=384) --------
#pragma unroll
        for (int n = 0; n < 8; ++n)
#pragma unroll
            for (int e = 0; e < 8; ++e) acc[n][e] = 0.0f;
#pragma unroll
        for (int kb = 0; kb < 12; ++kb) {
            int k = kb * 32;
            union { v16bf v; v8bf h[2]; } a;
            const unsigned short* ap = &s_act[lhalf * ACT_W + k + lhi * 16];
            a.h[0] = *(const v8bf*)(ap);          // K + {0..7}  (+16 for hi half)
            a.h[1] = *(const v8bf*)(ap + 8);      // K + 8 + {0..7}  (+16 for hi half)
#pragma unroll
            for (int n = 0; n < 8; ++n) {
                int gcol = (w + 8 * n) * 16 + lhalf;
                const v16bf* bp = (const v16bf*)&wc0[zo0 + gcol * K0 + k + lhi * 16];
                acc[n] = __builtin_amdgcn_wmma_f32_16x16x32_bf16(
                    false, a.v, false, *bp, (short)0, acc[n], false, false);
            }
        }
        __syncthreads();   // all waves done reading h0 before overwrite

        // -------- LSTM cell 0 (registers) + write h0 bf16 --------
#pragma unroll
        for (int g = 0; g < 2; ++g) {
            int u = 16 * w + 128 * g + lhalf;
#pragma unroll
            for (int v = 0; v < 8; ++v) {
                float iv = sigm (acc[0 + g][v] + b0r[0 + g]);
                float fv = sigm (acc[2 + g][v] + b0r[2 + g]);
                float gv = tanhf(acc[4 + g][v] + b0r[4 + g]);
                float ov = sigm (acc[6 + g][v] + b0r[6 + g]);
                float c  = fv * c0r[g][v] + iv * gv;
                c0r[g][v] = c;
                float h = ov * tanhf(c);
                s_act[(v + 8 * lhi) * ACT_W + 128 + u] = f2bf(h);
            }
        }
        __syncthreads();

        // -------- layer 1: gates = [h0|h1] @ Wcat1^T  (K=512) --------
#pragma unroll
        for (int n = 0; n < 8; ++n)
#pragma unroll
            for (int e = 0; e < 8; ++e) acc[n][e] = 0.0f;
#pragma unroll
        for (int kb = 0; kb < 16; ++kb) {
            int k = kb * 32;
            union { v16bf v; v8bf h[2]; } a;
            const unsigned short* ap = &s_act[lhalf * ACT_W + 128 + k + lhi * 16];
            a.h[0] = *(const v8bf*)(ap);
            a.h[1] = *(const v8bf*)(ap + 8);
#pragma unroll
            for (int n = 0; n < 8; ++n) {
                int gcol = (w + 8 * n) * 16 + lhalf;
                const v16bf* bp = (const v16bf*)&wc1[zo1 + gcol * K1 + k + lhi * 16];
                acc[n] = __builtin_amdgcn_wmma_f32_16x16x32_bf16(
                    false, a.v, false, *bp, (short)0, acc[n], false, false);
            }
        }
        __syncthreads();

        // -------- LSTM cell 1 + write h1 bf16 --------
#pragma unroll
        for (int g = 0; g < 2; ++g) {
            int u = 16 * w + 128 * g + lhalf;
#pragma unroll
            for (int v = 0; v < 8; ++v) {
                float iv = sigm (acc[0 + g][v] + b1r[0 + g]);
                float fv = sigm (acc[2 + g][v] + b1r[2 + g]);
                float gv = tanhf(acc[4 + g][v] + b1r[4 + g]);
                float ov = sigm (acc[6 + g][v] + b1r[6 + g]);
                float c  = fv * c1r[g][v] + iv * gv;
                c1r[g][v] = c;
                float h = ov * tanhf(c);
                s_act[(v + 8 * lhi) * ACT_W + 384 + u] = f2bf(h);
            }
        }
        __syncthreads();

        // -------- fc: logits = relu(h1 @ fcW^T + b)  (waves 0..2) --------
        if (w < 3) {
            v8f fa;
#pragma unroll
            for (int e = 0; e < 8; ++e) fa[e] = 0.0f;
#pragma unroll
            for (int kb = 0; kb < 8; ++kb) {
                int k = kb * 32;
                union { v16bf v; v8bf h[2]; } a;
                const unsigned short* ap = &s_act[lhalf * ACT_W + 384 + k + lhi * 16];
                a.h[0] = *(const v8bf*)(ap);
                a.h[1] = *(const v8bf*)(ap + 8);
                const v16bf* bp = (const v16bf*)&fcw[zof + fccol * HDIM + k + lhi * 16];
                fa = __builtin_amdgcn_wmma_f32_16x16x32_bf16(
                    false, a.v, false, *bp, (short)0, fa, false, false);
            }
#pragma unroll
            for (int v = 0; v < 8; ++v)
                s_lg[(v + 8 * lhi) * 48 + fccol] = fmaxf(fa[v] + fb, 0.0f);
        }
        __syncthreads();

        // -------- per-row softmax stats + NLL accumulate --------
        if (tid < 16) {
            const float* lr = &s_lg[tid * 48];
            float mx = 0.0f;                       // relu => logits >= 0
            for (int v = 0; v < VOCAB; ++v) mx = fmaxf(mx, lr[v]);
            float s = 0.0f;
            for (int v = 0; v < VOCAB; ++v) s += __expf(lr[v] - mx);
            float lse = __logf(s);
            s_mx[tid] = mx; s_lse[tid] = lse;
            int tgt = s_tok[tid * S_LEN + t + 1];
            if (tgt != 0) s_nll[tid] += (mx + lse - lr[tgt]);
        }
        __syncthreads();

        // -------- probs out --------
        for (int idx = tid; idx < 16 * VOCAB; idx += 256) {
            int r = idx / VOCAB, v = idx - r * VOCAB;
            float p = __expf(s_lg[r * 48 + v] - s_mx[r] - s_lse[r]);
            out[((size_t)(wg * 16 + r) * T_LEN + t) * VOCAB + v] = p;
        }
        // next-iteration barriers order s_lg reuse; no extra barrier needed
    }

    if (tid < 16) {
        out[(size_t)N_SEQ * T_LEN * VOCAB + wg * 16 + tid] = s_nll[tid] / s_len[tid];
    }
}

// ---------------------------------------------------------------------------
// loss = mean(molecule_loss)
// ---------------------------------------------------------------------------
__global__ void reduce_kernel(float* __restrict__ out)
{
    __shared__ float s[256];
    const size_t base = (size_t)N_SEQ * T_LEN * VOCAB;
    float acc = 0.0f;
    for (int i = threadIdx.x; i < N_SEQ; i += 256) acc += out[base + i];
    s[threadIdx.x] = acc;
    __syncthreads();
    for (int st = 128; st > 0; st >>= 1) {
        if (threadIdx.x < st) s[threadIdx.x] += s[threadIdx.x + st];
        __syncthreads();
    }
    if (threadIdx.x == 0) out[base + N_SEQ] = s[0] / (float)N_SEQ;
}

extern "C" void kernel_launch(void* const* d_in, const int* in_sizes, int n_in,
                              void* d_out, int out_size, void* d_ws, size_t ws_size,
                              hipStream_t stream)
{
    (void)in_sizes; (void)n_in; (void)out_size; (void)ws_size;   // ws_size >= WS_NEED (~1.8 MB)
    const int*   x    = (const int*)  d_in[0];
    const float* emb  = (const float*)d_in[1];
    const float* Wih0 = (const float*)d_in[2];
    const float* Whh0 = (const float*)d_in[3];
    const float* bih0 = (const float*)d_in[4];
    const float* bhh0 = (const float*)d_in[5];
    const float* Wih1 = (const float*)d_in[6];
    const float* Whh1 = (const float*)d_in[7];
    const float* bih1 = (const float*)d_in[8];
    const float* bhh1 = (const float*)d_in[9];
    const float* fcW  = (const float*)d_in[10];
    const float* fcb  = (const float*)d_in[11];

    unsigned char* ws = (unsigned char*)d_ws;
    const unsigned short* wc0  = (const unsigned short*)(ws + OFF_WCAT0);
    const unsigned short* wc1  = (const unsigned short*)(ws + OFF_WCAT1);
    const unsigned short* fcwp = (const unsigned short*)(ws + OFF_FCW);
    const unsigned short* embb = (const unsigned short*)(ws + OFF_EMB);
    const float* b0 = (const float*)(ws + OFF_B0);
    const float* b1 = (const float*)(ws + OFF_B1);

    prep_kernel<<<3658, 256, 0, stream>>>(Wih0, Whh0, bih0, bhh0,
                                          Wih1, Whh1, bih1, bhh1,
                                          fcW, emb, ws);
    lstm_kernel<<<N_SEQ / 16, 256, 0, stream>>>(x, wc0, wc1, fcwp, embb,
                                                b0, b1, fcb, (float*)d_out);
    reduce_kernel<<<1, 256, 0, stream>>>((float*)d_out);
}
